// SelfAttention_21784074125405
// MI455X (gfx1250) — matrix operand order repaired and checked by reading, not compiled, not run on hardware
//
#include <hip/hip_runtime.h>
#include <hip/hip_bf16.h>

typedef _Float16 half_t;
typedef __attribute__((ext_vector_type(16))) _Float16 v16h;
typedef __attribute__((ext_vector_type(8)))  _Float16 v8h;   // 16B aligned
typedef __attribute__((ext_vector_type(8)))  float    v8f;

namespace {
constexpr int kB = 2, kD = 1024, kL = 2048, kH = 16, kDK = 64;
constexpr int kE = kH * kDK;               // 1024
constexpr float kNegInf = -1e30f;
constexpr int kWaves = 8;                  // 256 threads / wave32
constexpr int kRowsPerBlock = kWaves * 16; // 128
}

__device__ __forceinline__ int lane_id() { return (int)(threadIdx.x & 31u); }

__device__ __forceinline__ v16h cat16(v8h lo, v8h hi) {
  return __builtin_shufflevector(lo, hi, 0, 1, 2, 3, 4, 5, 6, 7,
                                 8, 9, 10, 11, 12, 13, 14, 15);
}

// ---- WMMA fragment helpers (layouts per CDNA5 ISA §7.12.2) ----
// All loads are per-lane contiguous 16B chunks -> global/ds _load_b128.

// A fragment: 16x32 f16 tile from row-major source, leading dim lda (halves).
// lanes 0-15: row m=lane, slots 0..7 -> K0..7, 8..15 -> K16..23; lanes 16-31
// hold the K8..15 / K24..31 halves. Requires 16B alignment of base rows.
__device__ __forceinline__ v16h load_a_frag(const half_t* __restrict__ base, int lda) {
  const int lane = lane_id();
  const int m  = lane & 15;
  const int kb = (lane >> 4) << 3;         // 0 or 8
  const half_t* p = base + (size_t)m * lda + kb;
  v8h lo = *reinterpret_cast<const v8h*>(p);
  v8h hi = *reinterpret_cast<const v8h*>(p + 16);
  return cat16(lo, hi);
}

// B fragment from an N x K row-major source (i.e. b(k,n) = M[n][k]):
// slots 0..7 hold K = kb + j, slots 8..15 hold K = 16 + kb + (j-8), lane(0..15) = N.
__device__ __forceinline__ v16h load_bT_frag(const half_t* __restrict__ base, int ldm) {
  const int lane = lane_id();
  const int n  = lane & 15;
  const int kb = (lane >> 4) << 3;         // 0 or 8
  const half_t* p = base + (size_t)n * ldm + kb;
  const v8h* p8 = reinterpret_cast<const v8h*>(p);
  return cat16(p8[0], p8[2]);
}

// C/D layout: VGPR r -> row r (lanes 0-15) / row r+8 (lanes 16-31), N = lane&15.
// Row-major strided f16 store (for Q, K, heads).
__device__ __forceinline__ void store_c_f16(half_t* __restrict__ base, int ldd, v8f c) {
  const int lane = lane_id();
  const int n  = lane & 15;
  const int mo = (lane >> 4) << 3;
  half_t* p = base + (size_t)mo * ldd + n;
#pragma unroll
  for (int r = 0; r < 8; ++r) p[(size_t)r * ldd] = (half_t)c[r];
}

// Transposed f16 store: dst[col*ldd + row]. Per-lane contiguous 16B -> b128 store.
__device__ __forceinline__ void store_c_f16_T(half_t* __restrict__ base, int ldd, v8f c) {
  const int lane = lane_id();
  const int n  = lane & 15;
  const int mo = (lane >> 4) << 3;
  half_t* p = base + (size_t)n * ldd + mo;
  v8h h;
#pragma unroll
  for (int r = 0; r < 8; ++r) h[r] = (half_t)c[r];
  *reinterpret_cast<v8h*>(p) = h;
}

__device__ __forceinline__ v8f wmma_f16(v16h a, v16h b, v8f c) {
  return __builtin_amdgcn_wmma_f32_16x16x32_f16(false, a, false, b, (short)0, c,
                                                false, false);
}

// ---- Kernel 1: batched transpose + f32->f16 convert ----
// src: (batch, rows, cols) f32 ; dst: (batch, cols, rows) f16, output-contiguous.
__global__ __launch_bounds__(256) void cvtT_kernel(const float* __restrict__ src,
                                                   half_t* __restrict__ dst,
                                                   int rows, int cols) {
  const int bidx = blockIdx.y;
  const int idx = blockIdx.x * 256 + threadIdx.x;
  if (idx >= rows * cols) return;
  const int r = idx % rows;
  const int c = idx / rows;
  dst[(size_t)bidx * rows * cols + idx] =
      (half_t)src[(size_t)bidx * rows * cols + (size_t)r * cols + c];
}

// ---- Kernel 2: QKV projection. grid (L/128, H, B), 256 thr. ----
// Q/K[b,h] (LxDK, row-major) and V^T[b,h] (DKxL) from xh (B,L,D) @ W^T (H,DK,D).
__global__ __launch_bounds__(256) void qkv_kernel(const half_t* __restrict__ xh,
    const half_t* __restrict__ wqt, const half_t* __restrict__ wkt,
    const half_t* __restrict__ wvt,
    half_t* __restrict__ q, half_t* __restrict__ k, half_t* __restrict__ vT) {
  const int wid = (int)(threadIdx.x >> 5);
  const int b = blockIdx.z, h = blockIdx.y;
  const int row0 = blockIdx.x * kRowsPerBlock + wid * 16;
  const half_t* xbase = xh + (size_t)(b * kL + row0) * kD;
  const half_t* wlist[3] = {wqt, wkt, wvt};
#pragma unroll
  for (int which = 0; which < 3; ++which) {
    const half_t* wbase = wlist[which] + (size_t)h * kDK * kD;
    v8f acc[4] = {};
    for (int kk = 0; kk < kD; kk += 32) {
      v16h a = load_a_frag(xbase + kk, kD);
#pragma unroll
      for (int t = 0; t < 4; ++t) {
        // b(k,n) = W[kk+k][t*16+n] = Wt[t*16+n][kk+k] -> contiguous bT load
        v16h bf = load_bT_frag(wbase + (size_t)(t * 16) * kD + kk, kD);
        acc[t] = wmma_f16(a, bf, acc[t]);
      }
    }
    if (which == 0) {
      half_t* obase = q + (size_t)((b * kH + h) * kL + row0) * kDK;
#pragma unroll
      for (int t = 0; t < 4; ++t) store_c_f16(obase + t * 16, kDK, acc[t]);
    } else if (which == 1) {
      half_t* obase = k + (size_t)((b * kH + h) * kL + row0) * kDK;
#pragma unroll
      for (int t = 0; t < 4; ++t) store_c_f16(obase + t * 16, kDK, acc[t]);
    } else {
      // vT[b,h][dk][l]: transposed store, per-lane contiguous b128
      half_t* obase = vT + (size_t)(b * kH + h) * kDK * kL + row0;
#pragma unroll
      for (int t = 0; t < 4; ++t) store_c_f16_T(obase + (size_t)(t * 16) * kL, kL, acc[t]);
    }
  }
}

// ---- Kernel 3: flash attention. grid (L/128, H, B), 256 thr. ----
__global__ __launch_bounds__(256) void attn_kernel(const half_t* __restrict__ q,
    const half_t* __restrict__ kmat, const half_t* __restrict__ vT,
    const float* __restrict__ mask, half_t* __restrict__ heads) {
  __shared__ alignas(16) half_t lds_p[kWaves][16][72];  // 144B row pitch
  const int wid  = (int)(threadIdx.x >> 5);
  const int lane = lane_id();
  const int n    = lane & 15;
  const int hi   = lane >> 4;
  const int b = blockIdx.z, h = blockIdx.y;
  const int qrow0 = blockIdx.x * kRowsPerBlock + wid * 16;

  const half_t* qbase  = q    + (size_t)((b * kH + h) * kL + qrow0) * kDK;
  const half_t* kbase  = kmat + (size_t)((b * kH + h) * kL) * kDK;
  const half_t* vTbase = vT   + (size_t)(b * kH + h) * kDK * kL;
  const float*  mrow   = mask + (size_t)b * kL;

  v16h qa0 = load_a_frag(qbase, kDK);
  v16h qa1 = load_a_frag(qbase + 32, kDK);

  v8f o[4] = {};
  float Mr[8], Lr[8];
#pragma unroll
  for (int r = 0; r < 8; ++r) { Mr[r] = kNegInf; Lr[r] = 0.0f; }
  const float scale = 0.125f;  // 1/sqrt(DK)

  for (int key0 = 0; key0 < kL; key0 += 64) {
    if (key0 + 64 < kL) {  // stream next K/V chunk into cache
      __builtin_prefetch(kbase + (size_t)(key0 + 64) * kDK, 0, 1);
      __builtin_prefetch(vTbase + key0 + 64, 0, 1);
    }
    // S = Q * K^T : b(k=dk, n=key) = K[key][dk] -> contiguous bT load
    v8f s[4] = {};
#pragma unroll
    for (int t = 0; t < 4; ++t) {
      v16h b0 = load_bT_frag(kbase + (size_t)(key0 + t * 16) * kDK, kDK);
      v16h b1 = load_bT_frag(kbase + (size_t)(key0 + t * 16) * kDK + 32, kDK);
      s[t] = wmma_f16(qa0, b0, s[t]);
      s[t] = wmma_f16(qa1, b1, s[t]);
    }
    // key mask: masked = m*(s*scale) + (1-m)*NEG_INF
#pragma unroll
    for (int t = 0; t < 4; ++t) {
      float mv   = mrow[key0 + t * 16 + n];
      float bias = (1.0f - mv) * kNegInf;
#pragma unroll
      for (int r = 0; r < 8; ++r) s[t][r] = mv * (s[t][r] * scale) + bias;
    }
    // online softmax; row r of C tiles lives across a 16-lane group
#pragma unroll
    for (int r = 0; r < 8; ++r) {
      float mx = fmaxf(fmaxf(s[0][r], s[1][r]), fmaxf(s[2][r], s[3][r]));
#pragma unroll
      for (int dd = 1; dd < 16; dd <<= 1) mx = fmaxf(mx, __shfl_xor(mx, dd, 32));
      float Mn    = fmaxf(Mr[r], mx);
      float alpha = __expf(Mr[r] - Mn);
      float rs = 0.0f;
#pragma unroll
      for (int t = 0; t < 4; ++t) { s[t][r] = __expf(s[t][r] - Mn); rs += s[t][r]; }
#pragma unroll
      for (int dd = 1; dd < 16; dd <<= 1) rs += __shfl_xor(rs, dd, 32);
      Lr[r] = Lr[r] * alpha + rs;
      Mr[r] = Mn;
#pragma unroll
      for (int t = 0; t < 4; ++t) o[t][r] *= alpha;
    }
    // stage P (C layout) -> LDS -> A layout; DS ops are in-order per wave
#pragma unroll
    for (int t = 0; t < 4; ++t)
#pragma unroll
      for (int r = 0; r < 8; ++r)
        lds_p[wid][hi * 8 + r][t * 16 + n] = (half_t)s[t][r];
    __syncthreads();
    v16h pa0 = load_a_frag(&lds_p[wid][0][0], 72);
    v16h pa1 = load_a_frag(&lds_p[wid][0][0] + 32, 72);
    __syncthreads();
    // O += P * V : b(k=key, n=dk) = vT[dk][key] -> contiguous bT load
#pragma unroll
    for (int t = 0; t < 4; ++t) {
      v16h b0 = load_bT_frag(vTbase + (size_t)(t * 16) * kL + key0, kL);
      v16h b1 = load_bT_frag(vTbase + (size_t)(t * 16) * kL + key0 + 32, kL);
      o[t] = wmma_f16(pa0, b0, o[t]);
      o[t] = wmma_f16(pa1, b1, o[t]);
    }
  }
  // normalize, apply query-row mask, store heads (B, L, H*DK) f16
  half_t* hb = heads + (size_t)(b * kL + qrow0) * kE + h * kDK;
#pragma unroll
  for (int r = 0; r < 8; ++r) {
    int row = hi * 8 + r;
    float qm = mrow[qrow0 + row];
    float invl = qm / Lr[r];
#pragma unroll
    for (int t = 0; t < 4; ++t)
      hb[(size_t)row * kE + t * 16 + n] = (half_t)(o[t][r] * invl);
  }
}

// ---- Kernel 4: out = heads @ Wo (via Wo^T), stored transposed (B,1024,L) f32 ----
__global__ __launch_bounds__(256) void outproj_kernel(const half_t* __restrict__ heads,
    const half_t* __restrict__ woT, float* __restrict__ out) {
  const int wid  = (int)(threadIdx.x >> 5);
  const int lane = lane_id();
  const int n  = lane & 15;
  const int mo = (lane >> 4) << 3;
  const int b = blockIdx.z;
  const int col0 = blockIdx.y * 64;
  const int row0 = blockIdx.x * kRowsPerBlock + wid * 16;
  const half_t* abase = heads + (size_t)(b * kL + row0) * kE;
  v8f acc[4] = {};
  for (int kk = 0; kk < kE; kk += 32) {
    v16h a = load_a_frag(abase + kk, kE);
#pragma unroll
    for (int t = 0; t < 4; ++t) {
      // b(k,n) = Wo[kk+k][col0+n] = WoT[col0+n][kk+k] -> contiguous bT load
      v16h bf = load_bT_frag(woT + (size_t)(col0 + t * 16) * kE + kk, kE);
      acc[t] = wmma_f16(a, bf, acc[t]);
    }
  }
  // out[b][col][row]; contiguous in r -> vectorized f32 stores
#pragma unroll
  for (int t = 0; t < 4; ++t)
#pragma unroll
    for (int r = 0; r < 8; ++r)
      out[((size_t)b * kE + (col0 + t * 16 + n)) * kL + (row0 + mo + r)] = acc[t][r];
}

extern "C" void kernel_launch(void* const* d_in, const int* in_sizes, int n_in,
                              void* d_out, int out_size, void* d_ws, size_t ws_size,
                              hipStream_t stream) {
  (void)in_sizes; (void)n_in; (void)out_size; (void)ws_size;
  const float* x    = (const float*)d_in[0];
  const float* mask = (const float*)d_in[1];
  const float* Wq   = (const float*)d_in[2];
  const float* Wk   = (const float*)d_in[3];
  const float* Wv   = (const float*)d_in[4];
  const float* Wo   = (const float*)d_in[5];
  float* out = (float*)d_out;

  char* p = (char*)d_ws;
  auto take = [&](size_t bytes) {
    char* r = p; p += (bytes + 255) & ~(size_t)255; return r;
  };
  half_t* xh   = (half_t*)take((size_t)kB * kL * kD * 2);        // (B,L,D)    8 MB
  half_t* wqt  = (half_t*)take((size_t)kH * kDK * kD * 2);       // (H,DK,D)   2 MB
  half_t* wkt  = (half_t*)take((size_t)kH * kDK * kD * 2);
  half_t* wvt  = (half_t*)take((size_t)kH * kDK * kD * 2);
  half_t* woT  = (half_t*)take((size_t)kE * kE * 2);             // (N,K)      2 MB
  half_t* qh   = (half_t*)take((size_t)kB * kH * kL * kDK * 2);  // (B,H,L,DK) 8 MB
  half_t* kh   = (half_t*)take((size_t)kB * kH * kL * kDK * 2);
  half_t* vTh  = (half_t*)take((size_t)kB * kH * kDK * kL * 2);  // (B,H,DK,L)
  half_t* hh   = (half_t*)take((size_t)kB * kL * kE * 2);        // (B,L,E)    8 MB

  // transpose+convert: x (B,D,L)->(B,L,D); W (H,D,DK)->(H,DK,D); Wo -> Wo^T
  cvtT_kernel<<<dim3((kD * kL) / 256, kB), 256, 0, stream>>>(x, xh, kD, kL);
  cvtT_kernel<<<dim3((kD * kDK) / 256, kH), 256, 0, stream>>>(Wq, wqt, kD, kDK);
  cvtT_kernel<<<dim3((kD * kDK) / 256, kH), 256, 0, stream>>>(Wk, wkt, kD, kDK);
  cvtT_kernel<<<dim3((kD * kDK) / 256, kH), 256, 0, stream>>>(Wv, wvt, kD, kDK);
  cvtT_kernel<<<dim3((kE * kE) / 256, 1), 256, 0, stream>>>(Wo, woT, kE, kE);

  dim3 g(kL / kRowsPerBlock, kH, kB);       // (16, 16, 2)
  qkv_kernel<<<g, 256, 0, stream>>>(xh, wqt, wkt, wvt, qh, kh, vTh);
  attn_kernel<<<g, 256, 0, stream>>>(qh, kh, vTh, mask, hh);

  dim3 go(kL / kRowsPerBlock, kE / 64, kB); // (16, 16, 2)
  outproj_kernel<<<go, 256, 0, stream>>>(hh, woT, out);
}
